// FrozenSoftmaxAttention_21543555957302
// MI455X (gfx1250) — matrix operand order repaired and checked by reading, not compiled, not run on hardware
//
#include <hip/hip_runtime.h>
#include <hip/hip_bf16.h>
#include <math.h>
#include <stdint.h>

typedef __attribute__((ext_vector_type(16))) _Float16 v16h;
typedef __attribute__((ext_vector_type(8)))  _Float16 v8h;
typedef __attribute__((ext_vector_type(8)))  float    v8f;

#define WMMA_F32_F16(a, b, c) \
  __builtin_amdgcn_wmma_f32_16x16x32_f16(false, (a), false, (b), (short)0, (c), false, false)

// ---------- fragment loaders (layouts per CDNA5 ISA 7.12.2) ----------

// A fragment 16x32 f16 from row-major f32 (convert inline).
// lanes 0-15: M=lane, K in {0..7, 16..23}; lanes 16-31: M=lane-16, K in {8..15, 24..31}
static __device__ inline v16h load_a_f32(const float* __restrict__ base, int ld, int lane) {
  const int m  = lane & 15;
  const int kb = (lane >> 4) << 3;               // 0 or 8
  const float* p = base + (size_t)m * ld + kb;
  v16h a;
#pragma unroll
  for (int i = 0; i < 8; ++i) a[i] = (_Float16)p[i];
#pragma unroll
  for (int i = 0; i < 8; ++i) a[8 + i] = (_Float16)p[16 + i];
  return a;
}

// A fragment 16x32 f16 from row-major f16 (2x 128-bit loads per lane)
static __device__ inline v16h load_a_f16(const _Float16* base, int ld, int lane) {
  const int m  = lane & 15;
  const int kb = (lane >> 4) << 3;               // 0 or 8
  const _Float16* p = base + (size_t)m * ld + kb;
  v16h a;
  ((v8h*)&a)[0] = *(const v8h*)(p);              // K = kb .. kb+7
  ((v8h*)&a)[1] = *(const v8h*)(p + 16);         // K = kb+16 .. kb+23
  return a;
}

// B fragment 32x16 f16 loaded from BT (i.e. B transposed) row-major f32.
// lanes 0-15: N=lane, K in {0..7, 16..23}; lanes 16-31: N=lane-16, K in {8..15, 24..31}
static __device__ inline v16h load_bt_f32(const float* __restrict__ base, int ld, int lane) {
  const int n  = lane & 15;
  const int kb = (lane >> 4) << 3;               // 0 or 8
  const float* p = base + (size_t)n * ld + kb;
  v16h b;
#pragma unroll
  for (int i = 0; i < 8; ++i) b[i] = (_Float16)p[i];
#pragma unroll
  for (int i = 0; i < 8; ++i) b[8 + i] = (_Float16)p[16 + i];
  return b;
}

// B fragment from BT row-major f16 (2x 128-bit loads per lane)
static __device__ inline v16h load_bt_f16(const _Float16* base, int ld, int lane) {
  const int n  = lane & 15;
  const int kb = (lane >> 4) << 3;               // 0 or 8
  const _Float16* p = base + (size_t)n * ld + kb;
  v16h b;
  ((v8h*)&b)[0] = *(const v8h*)(p);
  ((v8h*)&b)[1] = *(const v8h*)(p + 16);
  return b;
}

// ---------------- K1: projection GEMM  Y = X(f32) * W(f32)^T  -> f16 ----------------
// M = 16384 (B*L), N = 512, K = 512. Grid 1024 blocks x 256 thr (8 waves).
// Each wave: 16x64 output tile. mode 0/1: row-major store; mode 2: store transposed
// per batch: Y[b][n][l]  (for V projection -> Vp^T, so PV B-operand is contiguous).
__global__ __launch_bounds__(256) void proj_gemm_kernel(
    const float* __restrict__ X, const float* __restrict__ W,
    _Float16* __restrict__ Y, int mode, float scaleOut)
{
  const int lane = threadIdx.x & 31;
  const int wv   = threadIdx.x >> 5;
  const int mt   = blockIdx.x * 16;
  const int n0   = wv * 64;

  v8f a0 = {}, a1 = {}, a2 = {}, a3 = {};
#pragma unroll 1
  for (int kf = 0; kf < 16; ++kf) {
    v16h a  = load_a_f32(X + (size_t)mt * 512 + kf * 32, 512, lane);
    v16h b0 = load_bt_f32(W + (size_t)(n0 +  0) * 512 + kf * 32, 512, lane);
    v16h b1 = load_bt_f32(W + (size_t)(n0 + 16) * 512 + kf * 32, 512, lane);
    v16h b2 = load_bt_f32(W + (size_t)(n0 + 32) * 512 + kf * 32, 512, lane);
    v16h b3 = load_bt_f32(W + (size_t)(n0 + 48) * 512 + kf * 32, 512, lane);
    a0 = WMMA_F32_F16(a, b0, a0);
    a1 = WMMA_F32_F16(a, b1, a1);
    a2 = WMMA_F32_F16(a, b2, a2);
    a3 = WMMA_F32_F16(a, b3, a3);
  }
  v8f acc[4] = {a0, a1, a2, a3};
  const int rbase = (lane < 16) ? 0 : 8;         // C layout: lanes 16-31 hold M=8..15
  const int nl = lane & 15;
#pragma unroll
  for (int nf = 0; nf < 4; ++nf) {
    const int col = n0 + nf * 16 + nl;
#pragma unroll
    for (int v = 0; v < 8; ++v) {
      const int m = mt + rbase + v;
      const float val = acc[nf][v] * scaleOut;
      if (mode < 2) {
        Y[(size_t)m * 512 + col] = (_Float16)val;
      } else {
        const int bb = m >> 12;                  // batch
        const int ll = m & 4095;                 // position
        Y[((size_t)bb * 512 + col) * 4096 + ll] = (_Float16)val;
      }
    }
  }
}

// ---------------- K2: flash attention, strict causal mask ----------------
// 1 block (8 waves, 256 thr) per 16-row query tile. grid = B * L/16 = 1024.
// Q tile staged into LDS via CDNA5 async global->LDS copies (ASYNCcnt).
// Per 128-wide key tile: wave w computes S cols [16w,16w+16) (16 WMMAs),
// cooperative online softmax in LDS, then wave w accumulates O cols [64w,64w+64)
// (16 WMMAs) from P(f16, LDS) x Vp^T(f16, global/L2).
__global__ __launch_bounds__(256) void flash_attn_kernel(
    const _Float16* __restrict__ Qh, const _Float16* __restrict__ Kh,
    const _Float16* __restrict__ VpT, _Float16* __restrict__ Oh)
{
  __shared__ __align__(16) _Float16 Qs[16 * 512];   // 16 KB
  __shared__ __align__(16) float    Sf[16 * 128];   // 8 KB
  __shared__ __align__(16) _Float16 Pf[16 * 128];   // 4 KB
  __shared__ float pmax[16][16];
  __shared__ float psum[16][16];
  __shared__ float rm[16], rl[16], rcorr[16], rnm[16];

  const int t    = threadIdx.x;
  const int lane = t & 31;
  const int wv   = t >> 5;
  const int b    = blockIdx.x >> 8;               // /256
  const int qb   = blockIdx.x & 255;              // query tile within batch
  const size_t bbase = (size_t)b * 4096;

  // stage Q tile (16 x 512 f16 = 1024 x 16B) into LDS with async DMA:
  // no VGPR round-trip; retired on ASYNCcnt.
  {
    const _Float16* src = Qh + (bbase + (size_t)qb * 16) * 512;
    const unsigned lds0 = (unsigned)(uintptr_t)(&Qs[0]);
#pragma unroll
    for (int rep = 0; rep < 4; ++rep) {
      const int i = t + rep * 256;                // chunk index 0..1023
      const unsigned lds = lds0 + (unsigned)i * 16u;
      const _Float16* g = src + (size_t)i * 8;
      asm volatile("global_load_async_to_lds_b128 %0, %1, off"
                   :: "v"(lds), "v"(g) : "memory");
    }
    asm volatile("s_wait_asynccnt 0" ::: "memory");
  }
  if (t < 16) { rm[t] = -1e30f; rl[t] = 0.f; }
  __syncthreads();

  v8f o0 = {}, o1 = {}, o2 = {}, o3 = {};
  const int rbase = (lane < 16) ? 0 : 8;
  const int nl = lane & 15;
  const int r = t >> 4;                           // softmax row owned by this thread
  const int q = t & 15;                           // 16 threads per row, 8 cols each

  const int ntiles = (16 * qb + 14) / 128 + 1;    // strict causal: last needed j = 16qb+14
  for (int kt = 0; kt < ntiles; ++kt) {
    const int tb = kt * 128;

    // warm L2->WGP path for the next key tile's K rows (global_prefetch_b8)
    if (kt + 1 < ntiles) {
      __builtin_prefetch(Kh + (bbase + (size_t)(tb + 128 + wv * 16 + nl)) * 512, 0, 0);
    }

    // ---- scores: S_w = Qtile * Kh[tb+16w .. )^T, accumulate f32 ----
    v8f s = {};
    const _Float16* kptr = Kh + (bbase + (size_t)(tb + wv * 16)) * 512;
#pragma unroll 1
    for (int kf = 0; kf < 16; ++kf) {
      v16h a  = load_a_f16(Qs + kf * 32, 512, lane);
      v16h kb = load_bt_f16(kptr + kf * 32, 512, lane);
      s = WMMA_F32_F16(a, kb, s);
    }
    const int col = wv * 16 + nl;
    const int jg  = tb + col;
#pragma unroll
    for (int v = 0; v < 8; ++v) {
      const int ig = qb * 16 + rbase + v;
      Sf[(rbase + v) * 128 + col] = (jg >= ig) ? -1e30f : s[v];  // strict causal
    }
    __syncthreads();

    // ---- online softmax over the 16 x 128 tile ----
    float mx = -1e30f;
#pragma unroll
    for (int c = 0; c < 8; ++c) mx = fmaxf(mx, Sf[r * 128 + q * 8 + c]);
    pmax[r][q] = mx;
    __syncthreads();
    if (q == 0) {
      float tm = pmax[r][0];
      for (int i = 1; i < 16; ++i) tm = fmaxf(tm, pmax[r][i]);
      const float nm = fmaxf(rm[r], tm);
      rcorr[r] = __expf(rm[r] - nm);
      rnm[r] = nm;
      rm[r]  = nm;
    }
    __syncthreads();
    {
      const float nm = rnm[r];
      float ps = 0.f;
#pragma unroll
      for (int c = 0; c < 8; ++c) {
        const float p = __expf(Sf[r * 128 + q * 8 + c] - nm);
        Pf[r * 128 + q * 8 + c] = (_Float16)p;
        ps += p;
      }
      psum[r][q] = ps;
    }
    __syncthreads();
    if (q == 0) {
      float ss = 0.f;
      for (int i = 0; i < 16; ++i) ss += psum[r][i];
      rl[r] = rl[r] * rcorr[r] + ss;              // rcorr stable until next iteration
    }

    // ---- rescale O by correction, accumulate P(16x128) * V(128x64) ----
    float cr[8];
#pragma unroll
    for (int v = 0; v < 8; ++v) cr[v] = rcorr[rbase + v];
#pragma unroll
    for (int v = 0; v < 8; ++v) { o0[v] *= cr[v]; o1[v] *= cr[v]; o2[v] *= cr[v]; o3[v] *= cr[v]; }

    const _Float16* vbase = VpT + ((size_t)b * 512 + wv * 64) * 4096 + tb;
#pragma unroll 1
    for (int kf = 0; kf < 4; ++kf) {
      v16h a   = load_a_f16(Pf + kf * 32, 128, lane);
      v16h vb0 = load_bt_f16(vbase + (size_t) 0 * 4096 + kf * 32, 4096, lane);
      v16h vb1 = load_bt_f16(vbase + (size_t)16 * 4096 + kf * 32, 4096, lane);
      v16h vb2 = load_bt_f16(vbase + (size_t)32 * 4096 + kf * 32, 4096, lane);
      v16h vb3 = load_bt_f16(vbase + (size_t)48 * 4096 + kf * 32, 4096, lane);
      o0 = WMMA_F32_F16(a, vb0, o0);
      o1 = WMMA_F32_F16(a, vb1, o1);
      o2 = WMMA_F32_F16(a, vb2, o2);
      o3 = WMMA_F32_F16(a, vb3, o3);
    }
    __syncthreads();                              // protect Sf/Pf/rcorr for next tile
  }

  // ---- epilogue: O /= l, zero row 0 (reference zeroes it post-softmax) ----
  v8f oo[4] = {o0, o1, o2, o3};
#pragma unroll
  for (int nf = 0; nf < 4; ++nf) {
    const int col = wv * 64 + nf * 16 + nl;
#pragma unroll
    for (int v = 0; v < 8; ++v) {
      const int rr = rbase + v;
      float ov = oo[nf][v] / rl[rr];
      if (qb == 0 && rr == 0) ov = 0.f;
      Oh[(bbase + (size_t)qb * 16 + rr) * 512 + col] = (_Float16)ov;
    }
  }
}

// ---------------- K3: output projection  Vhat = Oh(f16) * Wo(f32)^T -> f32 ----------------
__global__ __launch_bounds__(256) void out_gemm_kernel(
    const _Float16* __restrict__ Oh, const float* __restrict__ Wo,
    float* __restrict__ Y)
{
  const int lane = threadIdx.x & 31;
  const int wv   = threadIdx.x >> 5;
  const int mt   = blockIdx.x * 16;
  const int n0   = wv * 64;

  v8f a0 = {}, a1 = {}, a2 = {}, a3 = {};
#pragma unroll 1
  for (int kf = 0; kf < 16; ++kf) {
    v16h a  = load_a_f16(Oh + (size_t)mt * 512 + kf * 32, 512, lane);
    v16h b0 = load_bt_f32(Wo + (size_t)(n0 +  0) * 512 + kf * 32, 512, lane);
    v16h b1 = load_bt_f32(Wo + (size_t)(n0 + 16) * 512 + kf * 32, 512, lane);
    v16h b2 = load_bt_f32(Wo + (size_t)(n0 + 32) * 512 + kf * 32, 512, lane);
    v16h b3 = load_bt_f32(Wo + (size_t)(n0 + 48) * 512 + kf * 32, 512, lane);
    a0 = WMMA_F32_F16(a, b0, a0);
    a1 = WMMA_F32_F16(a, b1, a1);
    a2 = WMMA_F32_F16(a, b2, a2);
    a3 = WMMA_F32_F16(a, b3, a3);
  }
  v8f acc[4] = {a0, a1, a2, a3};
  const int rbase = (lane < 16) ? 0 : 8;
  const int nl = lane & 15;
#pragma unroll
  for (int nf = 0; nf < 4; ++nf) {
    const int col = n0 + nf * 16 + nl;
#pragma unroll
    for (int v = 0; v < 8; ++v) {
      Y[(size_t)(mt + rbase + v) * 512 + col] = acc[nf][v];
    }
  }
}

// ---------------- launch ----------------
extern "C" void kernel_launch(void* const* d_in, const int* in_sizes, int n_in,
                              void* d_out, int out_size, void* d_ws, size_t ws_size,
                              hipStream_t stream) {
  (void)in_sizes; (void)n_in; (void)out_size; (void)ws_size;
  const float* Kin = (const float*)d_in[0];   // [4,4096,512]
  const float* Vin = (const float*)d_in[1];   // [4,4096,512]
  const float* Wq  = (const float*)d_in[2];   // [512,512]
  const float* Wk  = (const float*)d_in[3];
  const float* Wv  = (const float*)d_in[4];
  const float* Wo  = (const float*)d_in[5];
  float* out = (float*)d_out;                 // [4,4096,512] f32

  const size_t NELEM = (size_t)4 * 4096 * 512;      // 8,388,608
  _Float16* Qh  = (_Float16*)d_ws;                  // 16 MB  (scale folded in)
  _Float16* Kh  = Qh  + NELEM;                      // 16 MB
  _Float16* VpT = Kh  + NELEM;                      // 16 MB  [b][512][4096] transposed
  _Float16* Oh  = VpT + NELEM;                      // 16 MB

  const float scale = 5.0f / sqrtf(512.0f);         // BETA / sqrt(dh)

  proj_gemm_kernel<<<1024, 256, 0, stream>>>(Kin, Wq, Qh, 0, scale);
  proj_gemm_kernel<<<1024, 256, 0, stream>>>(Kin, Wk, Kh, 0, 1.0f);
  proj_gemm_kernel<<<1024, 256, 0, stream>>>(Vin, Wv, VpT, 2, 1.0f);
  flash_attn_kernel<<<1024, 256, 0, stream>>>(Qh, Kh, VpT, Oh);
  out_gemm_kernel<<<1024, 256, 0, stream>>>(Oh, Wo, out);
}
